// BoundaryHRM_80805514707184
// MI455X (gfx1250) — matrix operand order repaired and checked by reading, not compiled, not run on hardware
//
#include <hip/hip_runtime.h>
#include <hip/hip_bf16.h>

typedef __bf16 bf16;
typedef __attribute__((ext_vector_type(16))) __bf16 v16bf;
typedef __attribute__((ext_vector_type(8)))  float  v8f;

static constexpr int Bn  = 32;
static constexpr int Tn  = 512;
static constexpr int En  = 256;
static constexpr int Hn  = 512;
static constexpr int Cn  = 256;
static constexpr int NCn = 256;

// ---------------- workspace layout (bytes) ----------------
static constexpr size_t O_XBF   = 0;                       static constexpr size_t S_XBF   = (size_t)Bn*Tn*En*2;
static constexpr size_t O_WIH0  = O_XBF  + S_XBF;          static constexpr size_t S_WIH0  = (size_t)3*Hn*En*2;
static constexpr size_t O_WHH0  = O_WIH0 + S_WIH0;         static constexpr size_t S_WHH0  = (size_t)3*Hn*Hn*2;
static constexpr size_t O_WIH1  = O_WHH0 + S_WHH0;         static constexpr size_t S_WIH1  = (size_t)3*Hn*Hn*2;
static constexpr size_t O_WHH1  = O_WIH1 + S_WIH1;         static constexpr size_t S_WHH1  = (size_t)3*Hn*Hn*2;
static constexpr size_t O_SPW   = O_WHH1 + S_WHH1;         static constexpr size_t S_SPW   = (size_t)Cn*Hn*2;
static constexpr size_t O_CWIH  = O_SPW  + S_SPW;          static constexpr size_t S_CWIH  = (size_t)3*Cn*Cn*2;
static constexpr size_t O_CWHH  = O_CWIH + S_CWIH;         static constexpr size_t S_CWHH  = (size_t)3*Cn*Cn*2;
// zero region 1: hidden-state double buffers (bf16)
static constexpr size_t O_H0    = O_CWHH + S_CWHH;         static constexpr size_t S_H0    = (size_t)2*Bn*Hn*2;
static constexpr size_t O_H1    = O_H0   + S_H0;           static constexpr size_t S_H1    = (size_t)2*Bn*Hn*2;
// rings
static constexpr size_t O_OUT0R = O_H1   + S_H1;           static constexpr size_t S_OUT0R = (size_t)4*Bn*Hn*2;
static constexpr size_t O_GOUTR = O_OUT0R+ S_OUT0R;        static constexpr size_t S_GOUTR = (size_t)4*Bn*Hn*4;
// zero region 2 (contiguous): halt-logit ring, seg_acc, halt_acc, ctrl f32, ctrl bf16, grid barrier
static constexpr size_t O_HLR   = O_GOUTR+ S_GOUTR;        static constexpr size_t S_HLR   = (size_t)4*Bn*4;
static constexpr size_t O_SEGA  = O_HLR  + S_HLR;          static constexpr size_t S_SEGA  = (size_t)Bn*Hn*4;
static constexpr size_t O_HACC  = O_SEGA + S_SEGA;         static constexpr size_t S_HACC  = (size_t)Bn*4;
static constexpr size_t O_CTRLF = O_HACC + S_HACC;         static constexpr size_t S_CTRLF = (size_t)Bn*Cn*4;
static constexpr size_t O_CTRLB = O_CTRLF+ S_CTRLF;        static constexpr size_t S_CTRLB = (size_t)Bn*Cn*2;
static constexpr size_t O_GBAR  = O_CTRLB+ S_CTRLB;        static constexpr size_t S_GBAR  = 64;
// outputs of persistent kernel + head scratch
static constexpr size_t O_CTRLO = O_GBAR + S_GBAR;         static constexpr size_t S_CTRLO = (size_t)Bn*Cn*4;
static constexpr size_t O_HT    = O_CTRLO+ S_CTRLO;        static constexpr size_t S_HT    = (size_t)Bn*Hn*4;
static constexpr size_t O_FUSED = O_HT   + S_HT;           static constexpr size_t S_FUSED = (size_t)Bn*Hn*4;
static constexpr size_t O_HDN   = O_FUSED+ S_FUSED;        static constexpr size_t S_HDN   = (size_t)Bn*Hn*4;

// ---------------- WMMA helpers ----------------
__device__ static inline v8f wmma_bf(v16bf a, v16bf b, v8f c) {
  return __builtin_amdgcn_wmma_f32_16x16x32_bf16(false, a, false, b, (short)0, c, false, false);
}

// A tile: 16 rows (rowbase..rowbase+15) x K32 window at k0.  Lane l15 = row, half selects K split.
__device__ static inline v16bf load_a(const bf16* A, int lda, int rowbase, int k0, int l15, int half) {
  union { v16bf v; uint4 q[2]; } u;
  const bf16* p = A + (size_t)(rowbase + l15) * lda + k0 + half * 8;
  u.q[0] = *(const uint4*)(p);
  u.q[1] = *(const uint4*)(p + 16);
  return u.v;
}
// B tile = W^T: output col = colbase+l15 is a contiguous row of W[N,K]; two 16-byte chunks / lane (k = half*8 and 16 + half*8).
__device__ static inline v16bf load_b(const bf16* W, int ldw, int colbase, int k0, int l15, int half) {
  union { v16bf v; uint4 q[2]; } u;
  const bf16* p = W + (size_t)(colbase + l15) * ldw + k0 + half * 8;
  u.q[0] = *(const uint4*)(p);
  u.q[1] = *(const uint4*)(p + 16);
  return u.v;
}

__device__ static inline float sigf(float x) { return 1.f / (1.f + __expf(-x)); }

// GRU gate accumulation for a 16-wide output column group:
// acc[0..1]=r (rt0,rt1), acc[2..3]=z, acc[4..5]=n_input, acc[6..7]=n_hidden (kept separate for r*h_n)
__device__ static void gru_accum(v8f* acc,
    const bf16* xA, int xlda, int Kx,
    const bf16* hA, int hdim,
    const bf16* wih, const bf16* whh,
    int colg, int l15, int half)
{
  for (int k = 0; k < Kx; k += 32) {
    v16bf a0 = load_a(xA, xlda, 0,  k, l15, half);
    v16bf a1 = load_a(xA, xlda, 16, k, l15, half);
    v16bf bR = load_b(wih, Kx, colg,          k, l15, half);
    v16bf bZ = load_b(wih, Kx, hdim + colg,   k, l15, half);
    v16bf bN = load_b(wih, Kx, 2*hdim + colg, k, l15, half);
    acc[0] = wmma_bf(a0, bR, acc[0]); acc[1] = wmma_bf(a1, bR, acc[1]);
    acc[2] = wmma_bf(a0, bZ, acc[2]); acc[3] = wmma_bf(a1, bZ, acc[3]);
    acc[4] = wmma_bf(a0, bN, acc[4]); acc[5] = wmma_bf(a1, bN, acc[5]);
  }
  for (int k = 0; k < hdim; k += 32) {
    v16bf a0 = load_a(hA, hdim, 0,  k, l15, half);
    v16bf a1 = load_a(hA, hdim, 16, k, l15, half);
    v16bf bR = load_b(whh, hdim, colg,          k, l15, half);
    v16bf bZ = load_b(whh, hdim, hdim + colg,   k, l15, half);
    v16bf bN = load_b(whh, hdim, 2*hdim + colg, k, l15, half);
    acc[0] = wmma_bf(a0, bR, acc[0]); acc[1] = wmma_bf(a1, bR, acc[1]);
    acc[2] = wmma_bf(a0, bZ, acc[2]); acc[3] = wmma_bf(a1, bZ, acc[3]);
    acc[6] = wmma_bf(a0, bN, acc[6]); acc[7] = wmma_bf(a1, bN, acc[7]);
  }
}

// One packed-GRU timestep for a wave's 16-column slice of h. h state lives in hst[] registers.
__device__ static void layer_step(int t,
    const bf16* xA, int xlda, int Kx,
    const bf16* hRd, bf16* hWr,
    const bf16* wih, const bf16* whh,
    const float* bih, const float* bhh,
    const int* lens, v8f* hst,
    bf16* outWr, float* goutWr,
    float* hlAcc, const float* bsw,
    int colg, int l15, int half)
{
  v8f acc[8] = {};
  gru_accum(acc, xA, xlda, Kx, hRd, Hn, wih, whh, colg, l15, half);
  const int col = colg + l15;
  const float bR  = bih[col] + bhh[col];
  const float bZ  = bih[Hn + col] + bhh[Hn + col];
  const float bNi = bih[2*Hn + col];
  const float bNh = bhh[2*Hn + col];
  const float bw  = bsw ? bsw[col] : 0.f;
  for (int rt = 0; rt < 2; ++rt) {
    for (int r = 0; r < 8; ++r) {
      const int row = rt*16 + r + 8*half;
      float rr = sigf(acc[0+rt][r] + bR);
      float zz = sigf(acc[2+rt][r] + bZ);
      float nn = tanhf(acc[4+rt][r] + bNi + rr * (acc[6+rt][r] + bNh));
      float hold = hst[rt][r];
      float hnew = (1.f - zz) * nn + zz * hold;
      bool  m    = t < lens[row];
      float hk   = m ? hnew : hold;          // freeze past length
      hst[rt][r] = hk;
      hWr[row*Hn + col] = (bf16)hk;          // broadcast frozen h for next step's A operand
      float ov = m ? hk : 0.f;               // pad_packed_sequence: zero padded outputs
      if (outWr)  outWr[row*Hn + col]  = (bf16)ov;
      if (goutWr) goutWr[row*Hn + col] = ov;
      if (hlAcc) {                            // partial halt logit: reduce 16 cols across half-wave
        float s = ov * bw;
        s += __shfl_xor(s, 1, 16);
        s += __shfl_xor(s, 2, 16);
        s += __shfl_xor(s, 4, 16);
        s += __shfl_xor(s, 8, 16);
        if (l15 == 0) atomicAdd(&hlAcc[row], s);
      }
    }
  }
}

// ---------------- grid barrier ----------------
struct GBar { unsigned cnt; unsigned gen; unsigned pad[14]; };
__device__ static void gbar(GBar* b, unsigned nb) {
  __syncthreads();
  if (threadIdx.x == 0) {
    __threadfence();
    unsigned g = __hip_atomic_load(&b->gen, __ATOMIC_RELAXED, __HIP_MEMORY_SCOPE_AGENT);
    if (atomicAdd(&b->cnt, 1u) == nb - 1u) {
      b->cnt = 0u;
      __threadfence();
      atomicAdd(&b->gen, 1u);
    } else {
      while (__hip_atomic_load(&b->gen, __ATOMIC_RELAXED, __HIP_MEMORY_SCOPE_AGENT) == g)
        __builtin_amdgcn_s_sleep(2);
      __threadfence();
    }
  }
  __syncthreads();
}

// ---------------- ACT controller step (one workgroup, wave32 x 8) ----------------
__device__ static void act_step(char* ws, int t2, bool finalmode,
    const int* lens, const float* bsb, const float* spb,
    const float* cbih, const float* cbhh,
    bf16* s_segrep, bf16* s_segin,
    float* s_ht, float* s_hdiv, int* s_fire,
    int lane, int wv)
{
  float* hlr   = (float*)(ws + O_HLR);
  float* sega  = (float*)(ws + O_SEGA);
  float* hacc  = (float*)(ws + O_HACC);
  float* ctrlf = (float*)(ws + O_CTRLF);
  bf16*  ctrlb = (bf16*) (ws + O_CTRLB);
  float* ctrlo = (float*)(ws + O_CTRLO);
  float* goutr = (float*)(ws + O_GOUTR);
  const bf16* spw  = (const bf16*)(ws + O_SPW);
  const bf16* cwih = (const bf16*)(ws + O_CWIH);
  const bf16* cwhh = (const bf16*)(ws + O_CWHH);
  const int l15 = lane & 15, half = lane >> 4;
  const int slot = t2 & 3;
  const int tid = threadIdx.x;

  // phase 1: halt prob, halt_acc update, fire decision
  if (tid < Bn) {
    const int b = tid;
    if (!finalmode) {
      float logit = hlr[slot*Bn + b] + bsb[0];
      bool valid = t2 < lens[b];
      if (!valid) logit = -20.f;
      float ht = sigf(logit);
      float ha = hacc[b] + ht;
      int fire = (ha >= 1.f) && valid;
      s_ht[b] = ht; s_hdiv[b] = fmaxf(ha, 1e-6f); s_fire[b] = fire;
      hacc[b] = fire ? ha - 1.f : ha;
      hlr[slot*Bn + b] = 0.f;                  // recycle ring slot
    } else {
      float ha = hacc[b];
      s_ht[b] = 0.f; s_hdiv[b] = fmaxf(ha, 1e-6f); s_fire[b] = (ha > 0.01f);
    }
  }
  __syncthreads();

  // phase 2: seg_acc += h_t * gru_out ; seg_repr = seg_acc / max(halt_acc,1e-6) ; reset on fire
  for (int i = tid; i < Bn*Hn; i += 256) {
    const int b = i >> 9;
    float sa = sega[i];
    if (!finalmode) {
      sa += s_ht[b] * goutr[(size_t)slot*Bn*Hn + i];
      sega[i] = s_fire[b] ? 0.f : sa;
    }
    s_segrep[i] = (bf16)(sa / s_hdiv[b]);
  }
  __syncthreads();

  // phase 3: seg_in = seg_repr @ sp_w^T + sp_b  (WMMA, 2 col groups / wave)
  for (int i = 0; i < 2; ++i) {
    const int cg = (wv*2 + i) * 16;
    v8f a0 = {}, a1 = {};
    for (int k = 0; k < Hn; k += 32) {
      v16bf A0 = load_a(s_segrep, Hn, 0,  k, l15, half);
      v16bf A1 = load_a(s_segrep, Hn, 16, k, l15, half);
      v16bf Bt = load_b(spw, Hn, cg, k, l15, half);
      a0 = wmma_bf(A0, Bt, a0); a1 = wmma_bf(A1, Bt, a1);
    }
    const float bias = spb[cg + l15];
    for (int rt = 0; rt < 2; ++rt)
      for (int r = 0; r < 8; ++r) {
        const int row = rt*16 + r + 8*half;
        s_segin[row*Cn + cg + l15] = (bf16)((rt ? a1[r] : a0[r]) + bias);
      }
  }
  __syncthreads();

  // phase 4: ctrl GRU gates (gi from seg_in, gh from ctrl) + fire-select, results kept in regs
  v8f sel[2][2];
  for (int i = 0; i < 2; ++i) {
    const int cg = (wv*2 + i) * 16;
    v8f acc[8] = {};
    gru_accum(acc, s_segin, Cn, Cn, ctrlb, Cn, cwih, cwhh, cg, l15, half);
    const int col = cg + l15;
    const float bR  = cbih[col] + cbhh[col];
    const float bZ  = cbih[Cn + col] + cbhh[Cn + col];
    const float bNi = cbih[2*Cn + col];
    const float bNh = cbhh[2*Cn + col];
    for (int rt = 0; rt < 2; ++rt)
      for (int r = 0; r < 8; ++r) {
        const int row = rt*16 + r + 8*half;
        float rr = sigf(acc[0+rt][r] + bR);
        float zz = sigf(acc[2+rt][r] + bZ);
        float nn = tanhf(acc[4+rt][r] + bNi + rr * (acc[6+rt][r] + bNh));
        float hold = ctrlf[row*Cn + col];
        float hnew = (1.f - zz) * nn + zz * hold;
        sel[i][rt][r] = s_fire[row] ? hnew : hold;
      }
  }
  __syncthreads();

  // phase 5: commit
  for (int i = 0; i < 2; ++i) {
    const int cg = (wv*2 + i) * 16;
    for (int rt = 0; rt < 2; ++rt)
      for (int r = 0; r < 8; ++r) {
        const int row = rt*16 + r + 8*half;
        const int col = cg + l15;
        float v = sel[i][rt][r];
        if (finalmode) ctrlo[row*Cn + col] = v;
        else { ctrlf[row*Cn + col] = v; ctrlb[row*Cn + col] = (bf16)v; }
      }
  }
}

// ---------------- persistent pipelined kernel ----------------
__global__ __launch_bounds__(256) void hrm_persistent(char* ws, const int* lens,
    const float* bih0, const float* bhh0,
    const float* bih1, const float* bhh1,
    const float* bsw,  const float* bsb,  const float* spb,
    const float* cbih, const float* cbhh)
{
  const bf16* xbf  = (const bf16*)(ws + O_XBF);
  const bf16* wih0 = (const bf16*)(ws + O_WIH0);
  const bf16* whh0 = (const bf16*)(ws + O_WHH0);
  const bf16* wih1 = (const bf16*)(ws + O_WIH1);
  const bf16* whh1 = (const bf16*)(ws + O_WHH1);
  bf16* h0buf = (bf16*)(ws + O_H0);
  bf16* h1buf = (bf16*)(ws + O_H1);
  bf16* out0r = (bf16*)(ws + O_OUT0R);
  float* goutr = (float*)(ws + O_GOUTR);
  float* hlr   = (float*)(ws + O_HLR);
  float* hT    = (float*)(ws + O_HT);
  GBar* bar = (GBar*)(ws + O_GBAR);

  __shared__ bf16  s_segrep[Bn*Hn];   // 32 KB
  __shared__ bf16  s_segin [Bn*Cn];   // 16 KB
  __shared__ float s_ht[Bn];
  __shared__ float s_hdiv[Bn];
  __shared__ int   s_fire[Bn];

  const int lane = threadIdx.x & 31;
  const int wv   = threadIdx.x >> 5;
  const int l15  = lane & 15;
  const int half = lane >> 4;
  const int role = (blockIdx.x < 4) ? 0 : (blockIdx.x < 8 ? 1 : 2);
  const int wg   = (role == 0) ? blockIdx.x*8 + wv : (blockIdx.x - 4)*8 + wv;
  const int colg = wg * 16;

  v8f hst[2] = {};   // this wave's 16-column slice of the layer hidden state

  for (int t = 0; t <= Tn + 2; ++t) {
    if (role == 0) {
      if (t < Tn) {
        layer_step(t, xbf + (size_t)t*En, Tn*En, En,
                   h0buf + (size_t)(t & 1)*Bn*Hn, h0buf + (size_t)((t+1) & 1)*Bn*Hn,
                   wih0, whh0, bih0, bhh0, lens, hst,
                   out0r + (size_t)(t & 3)*Bn*Hn, nullptr, nullptr, nullptr,
                   colg, l15, half);
      }
    } else if (role == 1) {
      if (t >= 1 && t <= Tn) {
        const int tt = t - 1;
        layer_step(tt, out0r + (size_t)(tt & 3)*Bn*Hn, Hn, Hn,
                   h1buf + (size_t)(tt & 1)*Bn*Hn, h1buf + (size_t)((tt+1) & 1)*Bn*Hn,
                   wih1, whh1, bih1, bhh1, lens, hst,
                   nullptr, goutr + (size_t)(tt & 3)*Bn*Hn, hlr + (size_t)(tt & 3)*Bn, bsw,
                   colg, l15, half);
        if (t == Tn) {  // frozen h_n of last layer
          for (int rt = 0; rt < 2; ++rt)
            for (int r = 0; r < 8; ++r) {
              const int row = rt*16 + r + 8*half;
              hT[row*Hn + colg + l15] = hst[rt][r];
            }
        }
      }
    } else {
      if (t >= 2) {
        if (t <= Tn + 1) act_step(ws, t - 2, false, lens, bsb, spb, cbih, cbhh,
                                  s_segrep, s_segin, s_ht, s_hdiv, s_fire, lane, wv);
        else             act_step(ws, 0,      true, lens, bsb, spb, cbih, cbhh,
                                  s_segrep, s_segin, s_ht, s_hdiv, s_fire, lane, wv);
      }
    }
    gbar(bar, 9u);
  }
}

// ---------------- prep kernels ----------------
__global__ void k_f32_to_bf16(const float* __restrict__ s, bf16* __restrict__ d, int n) {
  int i = blockIdx.x * 256 + threadIdx.x;
  if (i < n) d[i] = (bf16)s[i];
}
__global__ void k_zero32(float* p, int n) {
  int i = blockIdx.x * 256 + threadIdx.x;
  if (i < n) p[i] = 0.f;
}
__global__ void k_embed(const int* __restrict__ ids, const float* __restrict__ emb,
                        bf16* __restrict__ x, int n) {
  int i = blockIdx.x * 256 + threadIdx.x;
  if (i < n) {
    int e = i & (En - 1);
    int bt = i >> 8;               // En == 256
    x[i] = (bf16)emb[(size_t)ids[bt] * En + e];
  }
}

// ---------------- final gate + head (one-shot, exact f32) ----------------
__global__ __launch_bounds__(256) void k_head(
    const float* __restrict__ hT, const float* __restrict__ cT,
    const float* __restrict__ cow, const float* __restrict__ cob,
    const float* __restrict__ gw,  const float* __restrict__ gb,
    const float* __restrict__ hw1, const float* __restrict__ hb1,
    const float* __restrict__ hw2, const float* __restrict__ hb2,
    float* fused, float* hdn, float* out)
{
  __shared__ float s_gate[Bn];
  const int tid = threadIdx.x;
  if (tid < Bn) {
    float s = gb[0];
    for (int h = 0; h < Hn; ++h) s += hT[tid*Hn + h] * gw[h];
    for (int c = 0; c < Cn; ++c) s += cT[tid*Cn + c] * gw[Hn + c];
    s_gate[tid] = 1.f / (1.f + __expf(-s));
  }
  __syncthreads();
  for (int i = tid; i < Bn*Hn; i += 256) {
    int b = i >> 9, h = i & (Hn - 1);
    float p = cob[h];
    for (int c = 0; c < Cn; ++c) p += cT[b*Cn + c] * cow[h*Cn + c];
    fused[i] = hT[i] + s_gate[b] * p;
  }
  __syncthreads();
  for (int i = tid; i < Bn*Hn; i += 256) {
    int b = i >> 9, h = i & (Hn - 1);
    float s = hb1[h];
    for (int k = 0; k < Hn; ++k) s += fused[b*Hn + k] * hw1[h*Hn + k];
    hdn[i] = fmaxf(s, 0.f);
  }
  __syncthreads();
  for (int i = tid; i < Bn*NCn; i += 256) {
    int b = i >> 8, n = i & (NCn - 1);
    float s = hb2[n];
    for (int k = 0; k < Hn; ++k) s += hdn[b*Hn + k] * hw2[n*Hn + k];
    out[i] = s;
  }
}

// ---------------- host launcher ----------------
extern "C" void kernel_launch(void* const* d_in, const int* in_sizes, int n_in,
                              void* d_out, int out_size, void* d_ws, size_t ws_size,
                              hipStream_t stream) {
  char* ws = (char*)d_ws;
  const int*   ids   = (const int*)  d_in[0];
  const int*   lens  = (const int*)  d_in[1];
  const float* emb   = (const float*)d_in[2];
  const float* w_ih0 = (const float*)d_in[3];
  const float* w_hh0 = (const float*)d_in[4];
  const float* b_ih0 = (const float*)d_in[5];
  const float* b_hh0 = (const float*)d_in[6];
  const float* w_ih1 = (const float*)d_in[7];
  const float* w_hh1 = (const float*)d_in[8];
  const float* b_ih1 = (const float*)d_in[9];
  const float* b_hh1 = (const float*)d_in[10];
  const float* bs_w  = (const float*)d_in[11];
  const float* bs_b  = (const float*)d_in[12];
  const float* sp_w  = (const float*)d_in[13];
  const float* sp_b  = (const float*)d_in[14];
  const float* cwih  = (const float*)d_in[15];
  const float* cwhh  = (const float*)d_in[16];
  const float* cbih  = (const float*)d_in[17];
  const float* cbhh  = (const float*)d_in[18];
  const float* co_w  = (const float*)d_in[19];
  const float* co_b  = (const float*)d_in[20];
  const float* gw    = (const float*)d_in[21];
  const float* gb    = (const float*)d_in[22];
  const float* hw1   = (const float*)d_in[23];
  const float* hb1   = (const float*)d_in[24];
  const float* hw2   = (const float*)d_in[25];
  const float* hb2   = (const float*)d_in[26];

  auto cvt = [&](const float* src, size_t off, int n) {
    k_f32_to_bf16<<<(n + 255) / 256, 256, 0, stream>>>(src, (bf16*)(ws + off), n);
  };
  cvt(w_ih0, O_WIH0, 3*Hn*En);
  cvt(w_hh0, O_WHH0, 3*Hn*Hn);
  cvt(w_ih1, O_WIH1, 3*Hn*Hn);
  cvt(w_hh1, O_WHH1, 3*Hn*Hn);
  cvt(sp_w,  O_SPW,  Cn*Hn);
  cvt(cwih,  O_CWIH, 3*Cn*Cn);
  cvt(cwhh,  O_CWHH, 3*Cn*Cn);

  {
    int n = Bn*Tn*En;
    k_embed<<<(n + 255) / 256, 256, 0, stream>>>(ids, emb, (bf16*)(ws + O_XBF), n);
  }
  {
    int n1 = (int)((S_H0 + S_H1) / 4);   // h0/h1 double buffers
    k_zero32<<<(n1 + 255) / 256, 256, 0, stream>>>((float*)(ws + O_H0), n1);
    int n2 = (int)((S_HLR + S_SEGA + S_HACC + S_CTRLF + S_CTRLB + S_GBAR) / 4);
    k_zero32<<<(n2 + 255) / 256, 256, 0, stream>>>((float*)(ws + O_HLR), n2);
  }

  hrm_persistent<<<dim3(9), dim3(256), 0, stream>>>(ws, lens,
      b_ih0, b_hh0, b_ih1, b_hh1, bs_w, bs_b, sp_b, cbih, cbhh);

  k_head<<<dim3(1), dim3(256), 0, stream>>>(
      (const float*)(ws + O_HT), (const float*)(ws + O_CTRLO),
      co_w, co_b, gw, gb, hw1, hb1, hw2, hb2,
      (float*)(ws + O_FUSED), (float*)(ws + O_HDN), (float*)d_out);
}
